// choh_Decoder_SS2D_ViT_43654047596781
// MI455X (gfx1250) — matrix operand order repaired and checked by reading, not compiled, not run on hardware
//
#include <hip/hip_runtime.h>
#include <math.h>

typedef __attribute__((ext_vector_type(16))) _Float16 v16h;
typedef __attribute__((ext_vector_type(8)))  _Float16 v8h;
typedef __attribute__((ext_vector_type(8)))  float    v8f;

// ---------------- problem constants ----------------
#define B_    2
#define C_    32
#define H_    128
#define W_    128
#define L_    16384      // H*W
#define N_    16         // tokens
#define PD_   32768      // 32*32*32 patch dim
#define D_    512
#define HEADS_ 8
#define DHEAD_ 64
#define DI_   32
#define DS_   8
#define KDIR_ 4
#define OUTCH_ 20

// =====================================================================
// WMMA GEMM:  C[M,N] = act( A[M,K] @ Bw[K,N] + bias )
// Block tile 64x128, K-step 32, 8 waves (2x4), each wave 32x32 out
// (2x2 accumulators -> 4 v_wmma per wave per K-step).
// LDS: A row-major [m][k] (lane A-fragment = 2x ds_load_b128),
//      B transposed  [n][k] (lane B-fragment = 2x ds_load_b128).
// act: 0=none, 1=leaky_relu(0.2), 2=gelu(exact)
// =====================================================================
#define BM_ 64
#define BN_ 128
#define BK_ 32
#define LP_ 48   // LDS row pitch in halves (96B, keeps b128 loads aligned)

__global__ __launch_bounds__(256) void gemm_wmma_kernel(
    const float* __restrict__ A, const float* __restrict__ Bw,
    float* __restrict__ Cout, const float* __restrict__ bias,
    int M, int N, int K, int act)
{
    __shared__ _Float16 As[BM_][LP_];   // 6 KB
    __shared__ _Float16 Bs[BN_][LP_];   // 12 KB (transposed: [n][k])
    const int tid  = threadIdx.x;
    const int lane = tid & 31;
    const int wave = tid >> 5;
    const int wm = wave >> 2;          // 0..1  (32 rows each)
    const int wn = wave & 3;           // 0..3  (32 cols each)
    const int hi = lane >> 4;
    const int mr = lane & 15;
    const int block_m = blockIdx.y * BM_;
    const int block_n = blockIdx.x * BN_;

    v8f acc00 = {}, acc01 = {}, acc10 = {}, acc11 = {};

    const bool k_vec = ((K & 3) == 0);
    const bool n_vec = ((N & 3) == 0);

    for (int k0 = 0; k0 < K; k0 += BK_) {
        // ---- stage A tile 64x32 (row-major), 8 halves per thread ----
        {
            int r  = tid >> 2;            // 0..63
            int kc = (tid & 3) * 8;       // 0,8,16,24
            int gm = block_m + r;
            v8h hv;
            if (k_vec && gm < M && (k0 + kc + 8) <= K) {
                const float4 f0 = *(const float4*)&A[(size_t)gm * K + k0 + kc];
                const float4 f1 = *(const float4*)&A[(size_t)gm * K + k0 + kc + 4];
                hv[0] = (_Float16)f0.x; hv[1] = (_Float16)f0.y;
                hv[2] = (_Float16)f0.z; hv[3] = (_Float16)f0.w;
                hv[4] = (_Float16)f1.x; hv[5] = (_Float16)f1.y;
                hv[6] = (_Float16)f1.z; hv[7] = (_Float16)f1.w;
            } else {
#pragma unroll
                for (int u = 0; u < 8; ++u) {
                    int gk = k0 + kc + u;
                    hv[u] = (gm < M && gk < K) ? (_Float16)A[(size_t)gm * K + gk]
                                               : (_Float16)0.f;
                }
            }
            *(v8h*)&As[r][kc] = hv;       // 16B aligned (kc*2 in {0,16,32,48}, pitch 96)
        }
        // ---- stage B tile 32x128 transposed into [n][k], 16 halves/thread ----
        {
            int r0 = tid >> 4;            // 0..15
            int c8 = (tid & 15) * 8;      // 0..120
#pragma unroll
            for (int rr = r0; rr < BK_; rr += 16) {
                int gk = k0 + rr;
                _Float16 hv[8];
                if (n_vec && gk < K && (block_n + c8 + 8) <= N) {
                    const float4 f0 = *(const float4*)&Bw[(size_t)gk * N + block_n + c8];
                    const float4 f1 = *(const float4*)&Bw[(size_t)gk * N + block_n + c8 + 4];
                    hv[0] = (_Float16)f0.x; hv[1] = (_Float16)f0.y;
                    hv[2] = (_Float16)f0.z; hv[3] = (_Float16)f0.w;
                    hv[4] = (_Float16)f1.x; hv[5] = (_Float16)f1.y;
                    hv[6] = (_Float16)f1.z; hv[7] = (_Float16)f1.w;
                } else {
#pragma unroll
                    for (int u = 0; u < 8; ++u) {
                        int gn = block_n + c8 + u;
                        hv[u] = (gk < K && gn < N) ? (_Float16)Bw[(size_t)gk * N + gn]
                                                   : (_Float16)0.f;
                    }
                }
#pragma unroll
                for (int u = 0; u < 8; ++u) Bs[c8 + u][rr] = hv[u];
            }
        }
        // prefetch next K-tile while tiles are consumed
        if (k0 + BK_ < K) {
            if (tid < BM_ && (block_m + tid) < M)
                __builtin_prefetch(&A[(size_t)(block_m + tid) * K + k0 + BK_], 0, 0);
            if (tid >= 128 && tid < 128 + BK_ && (k0 + BK_ + tid - 128) < K)
                __builtin_prefetch(&Bw[(size_t)(k0 + BK_ + tid - 128) * N + block_n], 0, 0);
        }
        __syncthreads();

        // ---- fragments: contiguous b128 LDS loads ----
        // A lane layout: halves 0..7 = k[hi*8 .. +7], halves 8..15 = k[16+hi*8 .. +7]
        int ra0 = wm * 32 + mr;
        int ra1 = ra0 + 16;
        v8h a0l = *(const v8h*)&As[ra0][hi * 8];
        v8h a0h = *(const v8h*)&As[ra0][16 + hi * 8];
        v8h a1l = *(const v8h*)&As[ra1][hi * 8];
        v8h a1h = *(const v8h*)&As[ra1][16 + hi * 8];
        v16h af0 = __builtin_shufflevector(a0l, a0h, 0,1,2,3,4,5,6,7,8,9,10,11,12,13,14,15);
        v16h af1 = __builtin_shufflevector(a1l, a1h, 0,1,2,3,4,5,6,7,8,9,10,11,12,13,14,15);
        // B lane layout: halves 0..7 = k[hi*8 .. +7], halves 8..15 = k[16+hi*8 .. +7] at fixed column n
        int nc0 = wn * 32 + mr;
        int nc1 = nc0 + 16;
        v8h b0l = *(const v8h*)&Bs[nc0][hi * 8];
        v8h b0h = *(const v8h*)&Bs[nc0][16 + hi * 8];
        v8h b1l = *(const v8h*)&Bs[nc1][hi * 8];
        v8h b1h = *(const v8h*)&Bs[nc1][16 + hi * 8];
        v16h bf0 = __builtin_shufflevector(b0l, b0h, 0,1,2,3,4,5,6,7,8,9,10,11,12,13,14,15);
        v16h bf1 = __builtin_shufflevector(b1l, b1h, 0,1,2,3,4,5,6,7,8,9,10,11,12,13,14,15);

        acc00 = __builtin_amdgcn_wmma_f32_16x16x32_f16(false, af0, false, bf0, (short)0, acc00, false, false);
        acc01 = __builtin_amdgcn_wmma_f32_16x16x32_f16(false, af0, false, bf1, (short)0, acc01, false, false);
        acc10 = __builtin_amdgcn_wmma_f32_16x16x32_f16(false, af1, false, bf0, (short)0, acc10, false, false);
        acc11 = __builtin_amdgcn_wmma_f32_16x16x32_f16(false, af1, false, bf1, (short)0, acc11, false, false);
        __syncthreads();
    }

    // ---- store 2x2 subtiles ----
    v8f* accs[4] = { &acc00, &acc01, &acc10, &acc11 };
#pragma unroll
    for (int mi = 0; mi < 2; ++mi) {
#pragma unroll
        for (int ni = 0; ni < 2; ++ni) {
            v8f& a = *accs[mi * 2 + ni];
            int gm0 = block_m + wm * 32 + mi * 16 + hi * 8;
            int gn  = block_n + wn * 32 + ni * 16 + mr;
            if (gn < N) {
#pragma unroll
                for (int r = 0; r < 8; ++r) {
                    int gm = gm0 + r;
                    if (gm < M) {
                        float v = a[r];
                        if (bias) v += bias[gn];
                        if (act == 1)      v = v > 0.f ? v : 0.2f * v;
                        else if (act == 2) v = 0.5f * v * (1.0f + erff(v * 0.70710678118f));
                        Cout[(size_t)gm * N + gn] = v;
                    }
                }
            }
        }
    }
}

// =====================================================================
// LayerNorm over last dim, one block per row (in==out allowed)
// =====================================================================
__global__ __launch_bounds__(256) void ln_kernel(
    const float* __restrict__ in, float* __restrict__ out,
    const float* __restrict__ g, const float* __restrict__ bta, int dim)
{
    int row = blockIdx.x;
    const float* x = in + (size_t)row * dim;
    float* o = out + (size_t)row * dim;
    __shared__ float r1[256], r2[256];
    float s = 0.f, s2 = 0.f;
    for (int i = threadIdx.x; i < dim; i += 256) { float v = x[i]; s += v; s2 += v * v; }
    r1[threadIdx.x] = s; r2[threadIdx.x] = s2;
    __syncthreads();
    for (int off = 128; off > 0; off >>= 1) {
        if (threadIdx.x < off) { r1[threadIdx.x] += r1[threadIdx.x + off]; r2[threadIdx.x] += r2[threadIdx.x + off]; }
        __syncthreads();
    }
    float mean = r1[0] / dim;
    float var  = r2[0] / dim - mean * mean;
    float rstd = rsqrtf(var + 1e-5f);
    for (int i = threadIdx.x; i < dim; i += 256)
        o[i] = (x[i] - mean) * rstd * g[i] + bta[i];
}

// =====================================================================
// Small-N attention: one block per (b,head); N=16, DHEAD=64
// =====================================================================
__global__ __launch_bounds__(256) void attn_kernel(
    const float* __restrict__ qkv, float* __restrict__ out)
{
    int b = blockIdx.x >> 3;
    int h = blockIdx.x & 7;
    __shared__ float q[16][64], kk[16][64], vv[16][64], sc[16][16];
    int tid = threadIdx.x;
    for (int i = tid; i < 16 * 64; i += 256) {
        int n = i >> 6, d = i & 63;
        size_t base = (size_t)(b * 16 + n) * 1536 + h * 64 + d;
        q[n][d]  = qkv[base];
        kk[n][d] = qkv[base + 512];
        vv[n][d] = qkv[base + 1024];
    }
    __syncthreads();
    {
        int m = tid >> 4, j = tid & 15;
        float s = 0.f;
#pragma unroll
        for (int d = 0; d < 64; ++d) s += q[m][d] * kk[j][d];
        sc[m][j] = s * 0.125f;   // 1/sqrt(64)
    }
    __syncthreads();
    if (tid < 16) {
        float mx = -1e30f;
        for (int j = 0; j < 16; ++j) mx = fmaxf(mx, sc[tid][j]);
        float sum = 0.f;
        for (int j = 0; j < 16; ++j) { float e = expf(sc[tid][j] - mx); sc[tid][j] = e; sum += e; }
        float inv = 1.f / sum;
        for (int j = 0; j < 16; ++j) sc[tid][j] *= inv;
    }
    __syncthreads();
    for (int i = tid; i < 16 * 64; i += 256) {
        int m = i >> 6, d = i & 63;
        float o = 0.f;
#pragma unroll
        for (int j = 0; j < 16; ++j) o += sc[m][j] * vv[j][d];
        out[(size_t)(b * 16 + m) * 512 + h * 64 + d] = o;
    }
}

// =====================================================================
// Elementwise helpers
// =====================================================================
__global__ void fill0_kernel(float* p, int n)
{ int i = blockIdx.x * blockDim.x + threadIdx.x; if (i < n) p[i] = 0.f; }

__global__ void add_inplace_kernel(float* a, const float* b, int n)
{ int i = blockIdx.x * blockDim.x + threadIdx.x; if (i < n) a[i] += b[i]; }

__global__ void add_bcast_kernel(float* a, const float* b, int n, int mod)
{ int i = blockIdx.x * blockDim.x + threadIdx.x; if (i < n) a[i] += b[i % mod]; }

__global__ void resadd_lrelu_kernel(float* a, const float* b, int n)
{
    int i = blockIdx.x * blockDim.x + threadIdx.x;
    if (i < n) { float v = a[i] + b[i]; a[i] = v > 0.f ? v : 0.2f * v; }
}

// =====================================================================
// Patchify: in_imgs [B,C,H,W] -> pch [B*N, P*P*C]
// =====================================================================
__global__ void patchify_kernel(const float* __restrict__ img, float* __restrict__ pch)
{
    int idx = blockIdx.x * blockDim.x + threadIdx.x;   // B*N*PD = 1048576
    if (idx >= B_ * N_ * PD_) return;
    int j = idx & (PD_ - 1);
    int row = idx >> 15;
    int b = row >> 4, n = row & 15;
    int c  = j & 31;
    int px = (j >> 5) & 31;
    int py = j >> 10;
    int ph = n >> 2, pw = n & 3;
    int h = ph * 32 + py, w = pw * 32 + px;
    pch[idx] = img[((size_t)(b * C_ + c) << 14) + h * W_ + w];
}

// in_ksp [B,C,L] -> ksp_t [B*L, C]
__global__ void transpose_ksp_kernel(const float* __restrict__ ksp, float* __restrict__ out)
{
    int idx = blockIdx.x * blockDim.x + threadIdx.x;   // B*L*32
    if (idx >= B_ * L_ * C_) return;
    int c = idx & 31;
    int m = idx >> 5;
    int b = m >> 14, p = m & (L_ - 1);
    out[idx] = ksp[((size_t)(b * C_ + c) << 14) + p];
}

// depthwise 3x3 conv + bias + silu : xz[:, :32] -> xs0 [B,DI,H,W]
__global__ void dwconv_silu_kernel(const float* __restrict__ xz,
                                   const float* __restrict__ cw,
                                   const float* __restrict__ cb,
                                   float* __restrict__ xs0)
{
    int idx = blockIdx.x * blockDim.x + threadIdx.x;   // B*DI*L = 2^20
    if (idx >= B_ * DI_ * L_) return;
    int p = idx & (L_ - 1);
    int d = (idx >> 14) & 31;
    int b = idx >> 19;
    int h = p >> 7, w = p & 127;
    float acc = cb[d];
#pragma unroll
    for (int dy = 0; dy < 3; ++dy) {
#pragma unroll
        for (int dx = 0; dx < 3; ++dx) {
            int hh = h + dy - 1, ww = w + dx - 1;
            if (hh >= 0 && hh < H_ && ww >= 0 && ww < W_)
                acc += xz[(size_t)(b * L_ + hh * W_ + ww) * 64 + d] * cw[d * 9 + dy * 3 + dx];
        }
    }
    float sv = acc / (1.f + expf(-acc));   // silu
    xs0[((size_t)(b * DI_ + d) << 14) + p] = sv;
}

// scan-order pixel mapping for the 4 directions
__device__ __forceinline__ int pmap(int k, int l)
{
    switch (k) {
        case 0:  return l;
        case 1:  return (l & 127) * 128 + (l >> 7);
        case 2:  return (L_ - 1) - l;
        default: { int r = (L_ - 1) - l; return (r & 127) * 128 + (r >> 7); }
    }
}

// per (b,k,l): xproj -> (dtr, Bm, Cm), dts = softplus(dtr@dt_w + dt_b)
__global__ void ss_proj_kernel(const float* __restrict__ xs0,
                               const float* __restrict__ xp,
                               const float* __restrict__ dtw,
                               const float* __restrict__ dtb,
                               float* __restrict__ dts,
                               float* __restrict__ Bm,
                               float* __restrict__ Cm)
{
    int idx = blockIdx.x * blockDim.x + threadIdx.x;   // B*K*L = 131072
    if (idx >= B_ * KDIR_ * L_) return;
    int l = idx & (L_ - 1);
    int k = (idx >> 14) & 3;
    int b = idx >> 16;
    int p = pmap(k, l);
    float xv[DI_];
#pragma unroll
    for (int d = 0; d < DI_; ++d)
        xv[d] = xs0[((size_t)(b * DI_ + d) << 14) + p];

    float dtr0 = 0.f, dtr1 = 0.f;
#pragma unroll
    for (int d = 0; d < DI_; ++d) {
        const float* row = &xp[(size_t)(k * DI_ + d) * 18];
        dtr0 += xv[d] * row[0];
        dtr1 += xv[d] * row[1];
    }
    size_t bkl = (size_t)(b * KDIR_ + k) * L_ + l;
#pragma unroll
    for (int s = 0; s < DS_; ++s) {
        float bv = 0.f, cv = 0.f;
        for (int d = 0; d < DI_; ++d) {
            const float* row = &xp[(size_t)(k * DI_ + d) * 18];
            bv += xv[d] * row[2 + s];
            cv += xv[d] * row[10 + s];
        }
        Bm[bkl * 8 + s] = bv;
        Cm[bkl * 8 + s] = cv;
    }
    for (int d = 0; d < DI_; ++d) {
        float dt = dtr0 * dtw[(size_t)(k * 2 + 0) * DI_ + d]
                 + dtr1 * dtw[(size_t)(k * 2 + 1) * DI_ + d]
                 + dtb[k * DI_ + d];
        float sp = dt > 20.f ? dt : log1pf(expf(dt));
        dts[bkl * DI_ + d] = sp;
    }
}

// chunked associative scan: one block per (b,k,d); 256 threads x 64-chunks
__global__ __launch_bounds__(256) void ss_scan_kernel(
    const float* __restrict__ dts, const float* __restrict__ xs0,
    const float* __restrict__ Bm,  const float* __restrict__ Cm,
    const float* __restrict__ Alog, const float* __restrict__ Dp,
    float* __restrict__ y)
{
    int bkd = blockIdx.x;              // 256 total
    int d = bkd & 31;
    int k = (bkd >> 5) & 3;
    int b = bkd >> 7;
    int t = threadIdx.x;
    __shared__ float sA[256][DS_];
    __shared__ float sB[256][DS_];

    float Ac[DS_];
#pragma unroll
    for (int s = 0; s < DS_; ++s) Ac[s] = -expf(Alog[(size_t)(k * DI_ + d) * DS_ + s]);
    float Dval = Dp[k * DI_ + d];

    const int CH = L_ / 256;          // 64
    int l0 = t * CH;
    size_t bk = (size_t)(b * KDIR_ + k) * L_;

    float ap[DS_], bp[DS_];
#pragma unroll
    for (int s = 0; s < DS_; ++s) { ap[s] = 1.f; bp[s] = 0.f; }
    for (int i = 0; i < CH; ++i) {
        int l = l0 + i;
        int p = pmap(k, l);
        float dt = dts[(bk + l) * DI_ + d];
        float x  = xs0[((size_t)(b * DI_ + d) << 14) + p];
        const float* bm = &Bm[(bk + l) * 8];
        float dx = dt * x;
#pragma unroll
        for (int s = 0; s < DS_; ++s) {
            float a = expf(dt * Ac[s]);
            ap[s] *= a;
            bp[s]  = a * bp[s] + dx * bm[s];
        }
    }
#pragma unroll
    for (int s = 0; s < DS_; ++s) { sA[t][s] = ap[s]; sB[t][s] = bp[s]; }
    __syncthreads();
    if (t < DS_) {
        float hc = 0.f;
        for (int u = 0; u < 256; ++u) {
            float tmp = hc;
            hc = sA[u][t] * hc + sB[u][t];
            sB[u][t] = tmp;            // carry-in h for chunk u
        }
    }
    __syncthreads();
    float h[DS_];
#pragma unroll
    for (int s = 0; s < DS_; ++s) h[s] = sB[t][s];
    for (int i = 0; i < CH; ++i) {
        int l = l0 + i;
        int p = pmap(k, l);
        float dt = dts[(bk + l) * DI_ + d];
        float x  = xs0[((size_t)(b * DI_ + d) << 14) + p];
        const float* bm = &Bm[(bk + l) * 8];
        const float* cm = &Cm[(bk + l) * 8];
        float dx = dt * x;
        float yl = Dval * x;
#pragma unroll
        for (int s = 0; s < DS_; ++s) {
            float a = expf(dt * Ac[s]);
            h[s] = a * h[s] + dx * bm[s];
            yl  += h[s] * cm[s];
        }
        atomicAdd(&y[((size_t)b * L_ + p) * DI_ + d], yl);
    }
}

// LN over DI channels per pixel, * silu(z)
__global__ void ss_lngate_kernel(const float* __restrict__ y,
                                 const float* __restrict__ xz,
                                 const float* __restrict__ g,
                                 const float* __restrict__ bt,
                                 float* __restrict__ yg)
{
    int idx = blockIdx.x * blockDim.x + threadIdx.x;  // B*L
    if (idx >= B_ * L_) return;
    const float* yr = &y[(size_t)idx * DI_];
    float s = 0.f, s2 = 0.f;
#pragma unroll
    for (int d = 0; d < DI_; ++d) { float v = yr[d]; s += v; s2 += v * v; }
    float mean = s / DI_;
    float var  = s2 / DI_ - mean * mean;
    float rstd = rsqrtf(var + 1e-5f);
#pragma unroll
    for (int d = 0; d < DI_; ++d) {
        float v = (yr[d] - mean) * rstd * g[d] + bt[d];
        float z = xz[(size_t)idx * 64 + 32 + d];
        yg[(size_t)idx * DI_ + d] = v * (z / (1.f + expf(-z)));
    }
}

// concat -> xc NHWC [B,H,W,56]: [upsampled lat (4) | in_imgs (32) | out_ss (20)]
__global__ void concat_kernel(const float* __restrict__ lat,
                              const float* __restrict__ img,
                              const float* __restrict__ oss,
                              float* __restrict__ xc)
{
    int idx = blockIdx.x * blockDim.x + threadIdx.x;   // B*L*56
    if (idx >= B_ * L_ * 56) return;
    int c  = idx % 56;
    int m  = idx / 56;
    int b  = m >> 14, p = m & (L_ - 1);
    int h = p >> 7, w = p & 127;
    float v;
    if (c < 4) {
        int ph = h >> 5, fy = h & 31, pw = w >> 5, fx = w & 31;
        int n = ph * 4 + pw;
        v = lat[(size_t)(b * 16 + n) * 4096 + c * 1024 + fy * 32 + fx];
    } else if (c < 36) {
        v = img[((size_t)(b * C_ + (c - 4)) << 14) + p];
    } else {
        v = oss[(size_t)m * OUTCH_ + (c - 36)];
    }
    xc[idx] = v;
}

// im2col: src NHWC [B,H,W,Cin] -> col [B*L, Cin*9] (tap-major: (dy*3+dx)*Cin + c)
__global__ void im2col_kernel(const float* __restrict__ src, float* __restrict__ col,
                              int Cin)
{
    int Kc = Cin * 9;
    long long total = (long long)B_ * L_ * Kc;
    long long idx = (long long)blockIdx.x * blockDim.x + threadIdx.x;
    if (idx >= total) return;
    int kk = (int)(idx % Kc);
    int m  = (int)(idx / Kc);
    int tap = kk / Cin, c = kk % Cin;
    int dy = tap / 3 - 1, dx = tap % 3 - 1;
    int b = m >> 14, p = m & (L_ - 1);
    int h = (p >> 7) + dy, w = (p & 127) + dx;
    float v = 0.f;
    if (h >= 0 && h < H_ && w >= 0 && w < W_)
        v = src[((size_t)(b * L_ + h * W_ + w)) * Cin + c];
    col[idx] = v;
}

// repack OIHW conv weight -> [Cin*9, O] GEMM B-matrix
__global__ void repack_w_kernel(const float* __restrict__ w, float* __restrict__ out,
                                int O, int Cin)
{
    int idx = blockIdx.x * blockDim.x + threadIdx.x;   // 9*Cin*O
    if (idx >= 9 * Cin * O) return;
    int o = idx % O;
    int rest = idx / O;
    int c = rest % Cin;
    int tap = rest / Cin;
    out[idx] = w[((size_t)o * Cin + c) * 9 + tap];
}

// ---------------- FFT data-consistency stage ----------------
__device__ __forceinline__ float2 cadd2(float2 a, float2 b){ return make_float2(a.x+b.x, a.y+b.y); }
__device__ __forceinline__ float2 csub2(float2 a, float2 b){ return make_float2(a.x-b.x, a.y-b.y); }
__device__ __forceinline__ float2 cmul2(float2 a, float2 b){ return make_float2(a.x*b.x-a.y*b.y, a.x*b.y+a.y*b.x); }

__global__ void kprep_kernel(float2* __restrict__ kb, const float* __restrict__ sens,
                             const float* __restrict__ xri)
{
    int idx = blockIdx.x * blockDim.x + threadIdx.x;   // B*16*L = 524288
    if (idx >= B_ * 16 * L_) return;
    int p = idx & (L_ - 1);
    int coil = (idx >> 14) & 15;
    int b = idx >> 18;
    float sr = sens[((size_t)(b * C_ + 2 * coil)     << 14) + p];
    float si = sens[((size_t)(b * C_ + 2 * coil + 1) << 14) + p];
    float xr = xri[((size_t)(b * L_) + p) * 2 + 0];
    float xi = xri[((size_t)(b * L_) + p) * 2 + 1];
    kb[idx] = cmul2(make_float2(sr, si), make_float2(xr, xi));
}

// 128-pt radix-2 FFT per line with fftshift/ifftshift folded into indexing.
__global__ __launch_bounds__(128) void fft128_kernel(float2* __restrict__ data,
                                                     int axis, float dir)
{
    __shared__ float2 buf[128];
    int i = threadIdx.x;
    size_t base; int stride;
    if (axis == 0) { base = (size_t)blockIdx.x * 128; stride = 1; }
    else {
        int bc = blockIdx.x >> 7;
        int w  = blockIdx.x & 127;
        base = (size_t)bc * L_ + w; stride = 128;
    }
    int srcj = (i + 64) & 127;                        // ifftshift on load
    int r = (int)(__brev((unsigned)i) >> 25);         // bit-reversed placement
    buf[r] = data[base + (size_t)srcj * stride];
    __syncthreads();
    for (int s = 1; s <= 7; ++s) {
        int m = 1 << s, half = m >> 1;
        if (i < 64) {
            int grp = i / half, j = i % half;
            int pos = grp * m + j;
            float ang = dir * 6.283185307179586f * (float)j / (float)m;
            float2 wtw = make_float2(cosf(ang), sinf(ang));
            float2 u = buf[pos];
            float2 t = cmul2(wtw, buf[pos + half]);
            buf[pos]        = cadd2(u, t);
            buf[pos + half] = csub2(u, t);
        }
        __syncthreads();
    }
    float2 v = buf[(i + 64) & 127];                   // fftshift on store
    const float sc = 0.08838834764831845f;            // 1/sqrt(128), ortho per pass
    data[base + (size_t)i * stride] = make_float2(v.x * sc, v.y * sc);
}

__global__ void dc_kernel(float2* __restrict__ kb, const float* __restrict__ ksp,
                          const float* __restrict__ mask, const float* __restrict__ alpha)
{
    int idx = blockIdx.x * blockDim.x + threadIdx.x;
    if (idx >= B_ * 16 * L_) return;
    int p = idx & (L_ - 1);
    int coil = (idx >> 14) & 15;
    int b = idx >> 18;
    float mr = ksp[((size_t)(b * C_ + 2 * coil)     << 14) + p] * 100.f;
    float mi = ksp[((size_t)(b * C_ + 2 * coil + 1) << 14) + p] * 100.f;
    float ma = mask[(size_t)b * L_ + p] * alpha[0];
    float2 kp = kb[idx];
    kb[idx] = make_float2(kp.x + ma * (mr - kp.x), kp.y + ma * (mi - kp.y));
}

__global__ void combine_kernel(const float2* __restrict__ img,
                               const float* __restrict__ sens,
                               float* __restrict__ out)
{
    int idx = blockIdx.x * blockDim.x + threadIdx.x;   // B*L
    if (idx >= B_ * L_) return;
    int b = idx >> 14, p = idx & (L_ - 1);
    float ar = 0.f, ai = 0.f;
#pragma unroll
    for (int coil = 0; coil < 16; ++coil) {
        float2 v = img[((size_t)(b * 16 + coil) << 14) + p];
        float sr = sens[((size_t)(b * C_ + 2 * coil)     << 14) + p];
        float si = sens[((size_t)(b * C_ + 2 * coil + 1) << 14) + p];
        ar += v.x * sr + v.y * si;     // v * conj(s)
        ai += v.y * sr - v.x * si;
    }
    out[idx] = sqrtf(ar * ar + ai * ai + 1e-12f);
}

// =====================================================================
// Host orchestration
// =====================================================================
static inline dim3 gemm_grid(int M, int N) { return dim3((N + BN_ - 1) / BN_, (M + BM_ - 1) / BM_); }

extern "C" void kernel_launch(void* const* d_in, const int* in_sizes, int n_in,
                              void* d_out, int out_size, void* d_ws, size_t ws_size,
                              hipStream_t stream)
{
    (void)in_sizes; (void)n_in; (void)out_size; (void)ws_size;
    const float* in_imgs   = (const float*)d_in[0];
    const float* in_ksp    = (const float*)d_in[1];
    const float* maskp     = (const float*)d_in[2];
    const float* sens      = (const float*)d_in[3];
    const float* pe_ln1_g  = (const float*)d_in[4];
    const float* pe_ln1_b  = (const float*)d_in[5];
    const float* pe_w      = (const float*)d_in[6];
    const float* pe_b      = (const float*)d_in[7];
    const float* pe_ln2_g  = (const float*)d_in[8];
    const float* pe_ln2_b  = (const float*)d_in[9];
    const float* pos_emb   = (const float*)d_in[10];
    const float* enc_aln_g = (const float*)d_in[11];
    const float* enc_aln_b = (const float*)d_in[12];
    const float* enc_wqkv  = (const float*)d_in[13];
    const float* enc_wo    = (const float*)d_in[14];
    const float* enc_bo    = (const float*)d_in[15];
    const float* enc_fln_g = (const float*)d_in[16];
    const float* enc_fln_b = (const float*)d_in[17];
    const float* enc_w1    = (const float*)d_in[18];
    const float* enc_b1    = (const float*)d_in[19];
    const float* enc_w2    = (const float*)d_in[20];
    const float* enc_b2    = (const float*)d_in[21];
    const float* enc_ng    = (const float*)d_in[22];
    const float* enc_nb    = (const float*)d_in[23];
    const float* dec_pos   = (const float*)d_in[24];
    const float* dec_aln_g = (const float*)d_in[25];
    const float* dec_aln_b = (const float*)d_in[26];
    const float* dec_wqkv  = (const float*)d_in[27];
    const float* dec_wo    = (const float*)d_in[28];
    const float* dec_bo    = (const float*)d_in[29];
    const float* dec_fln_g = (const float*)d_in[30];
    const float* dec_fln_b = (const float*)d_in[31];
    const float* dec_w1    = (const float*)d_in[32];
    const float* dec_b1    = (const float*)d_in[33];
    const float* dec_w2    = (const float*)d_in[34];
    const float* dec_b2    = (const float*)d_in[35];
    const float* dec_ng    = (const float*)d_in[36];
    const float* dec_nb    = (const float*)d_in[37];
    const float* fin_w     = (const float*)d_in[38];
    const float* fin_b     = (const float*)d_in[39];
    const float* ss_in_w   = (const float*)d_in[40];
    const float* ss_in_b   = (const float*)d_in[41];
    const float* ss_conv_w = (const float*)d_in[42];
    const float* ss_conv_b = (const float*)d_in[43];
    const float* ss_xproj  = (const float*)d_in[44];
    const float* ss_dt_w   = (const float*)d_in[45];
    const float* ss_dt_b   = (const float*)d_in[46];
    const float* ss_Alog   = (const float*)d_in[47];
    const float* ss_D      = (const float*)d_in[48];
    const float* ss_ln_g   = (const float*)d_in[49];
    const float* ss_ln_b   = (const float*)d_in[50];
    const float* ss_out_w  = (const float*)d_in[51];
    const float* ss_out_b  = (const float*)d_in[52];
    const float* head_w    = (const float*)d_in[53];
    const float* head_b    = (const float*)d_in[54];
    const float* res_w1    = (const float*)d_in[55];
    const float* res_b1    = (const float*)d_in[56];
    const float* res_w2    = (const float*)d_in[57];
    const float* res_b2    = (const float*)d_in[58];
    const float* tail_w    = (const float*)d_in[59];
    const float* tail_b    = (const float*)d_in[60];
    const float* dc_alpha  = (const float*)d_in[61];
    float* out = (float*)d_out;

    // ---- workspace layout (16B-aligned float allocs) ----
    float* WS = (float*)d_ws;
    size_t off = 0;
    auto alloc = [&](size_t n) { float* p = WS + off; off += (n + 3) & ~(size_t)3; return p; };
    float* pch    = alloc((size_t)32 * PD_);
    float* pchln  = alloc((size_t)32 * PD_);
    float* tA     = alloc(32 * D_);
    float* tB     = alloc(32 * D_);
    float* tmp    = alloc(32 * D_);
    float* qkv    = alloc(32 * 3 * D_);
    float* attnout= alloc(32 * D_);
    float* ff1    = alloc(32 * 3072);
    float* lat    = alloc(32 * 4096);
    float* ksp_t  = alloc((size_t)B_ * L_ * C_);
    float* xz     = alloc((size_t)B_ * L_ * 64);
    float* xs0    = alloc((size_t)B_ * DI_ * L_);
    float* dts    = alloc((size_t)B_ * KDIR_ * L_ * DI_);
    float* Bmb    = alloc((size_t)B_ * KDIR_ * L_ * DS_);
    float* Cmb    = alloc((size_t)B_ * KDIR_ * L_ * DS_);
    float* ybuf   = alloc((size_t)B_ * L_ * DI_);
    float* ygate  = alloc((size_t)B_ * L_ * DI_);
    float* out_ss = alloc((size_t)B_ * L_ * OUTCH_);
    float* xc     = alloc((size_t)B_ * L_ * 56);
    float* colbuf = alloc((size_t)B_ * L_ * 576);
    float* wbuf   = alloc(576 * 64);
    float* h0     = alloc((size_t)B_ * L_ * 64);
    float* h1     = alloc((size_t)B_ * L_ * 64);
    float* h2     = alloc((size_t)B_ * L_ * 64);
    float* xri    = alloc((size_t)B_ * L_ * 2);
    float2* kbuf  = (float2*)alloc((size_t)B_ * 16 * L_ * 2);

    const int TPB = 256;
    auto blocks = [&](long long n) { return (unsigned)((n + TPB - 1) / TPB); };

    // ================= Patch embed + transformer =================
    patchify_kernel<<<blocks((long long)32 * PD_), TPB, 0, stream>>>(in_imgs, pch);
    ln_kernel<<<32, 256, 0, stream>>>(pch, pchln, pe_ln1_g, pe_ln1_b, PD_);
    gemm_wmma_kernel<<<gemm_grid(32, D_), 256, 0, stream>>>(pchln, pe_w, tmp, pe_b, 32, D_, PD_, 0);
    ln_kernel<<<32, 256, 0, stream>>>(tmp, tA, pe_ln2_g, pe_ln2_b, D_);
    add_bcast_kernel<<<blocks(32 * D_), TPB, 0, stream>>>(tA, pos_emb, 32 * D_, N_ * D_);

    for (int i = 0; i < 4; ++i) {
        const float* wqkv = enc_wqkv + (size_t)i * D_ * 3 * D_;
        const float* wo   = enc_wo   + (size_t)i * D_ * D_;
        const float* bo   = enc_bo   + (size_t)i * D_;
        const float* w1   = enc_w1   + (size_t)i * D_ * 2048;
        const float* b1   = enc_b1   + (size_t)i * 2048;
        const float* w2   = enc_w2   + (size_t)i * 2048 * D_;
        const float* b2   = enc_b2   + (size_t)i * D_;
        ln_kernel<<<32, 256, 0, stream>>>(tA, tB, enc_aln_g + i * D_, enc_aln_b + i * D_, D_);
        gemm_wmma_kernel<<<gemm_grid(32, 3 * D_), 256, 0, stream>>>(tB, wqkv, qkv, nullptr, 32, 3 * D_, D_, 0);
        attn_kernel<<<B_ * HEADS_, 256, 0, stream>>>(qkv, attnout);
        gemm_wmma_kernel<<<gemm_grid(32, D_), 256, 0, stream>>>(attnout, wo, tmp, bo, 32, D_, D_, 0);
        add_inplace_kernel<<<blocks(32 * D_), TPB, 0, stream>>>(tA, tmp, 32 * D_);
        ln_kernel<<<32, 256, 0, stream>>>(tA, tB, enc_fln_g + i * D_, enc_fln_b + i * D_, D_);
        gemm_wmma_kernel<<<gemm_grid(32, 2048), 256, 0, stream>>>(tB, w1, ff1, b1, 32, 2048, D_, 2);
        gemm_wmma_kernel<<<gemm_grid(32, D_), 256, 0, stream>>>(ff1, w2, tmp, b2, 32, D_, 2048, 0);
        add_inplace_kernel<<<blocks(32 * D_), TPB, 0, stream>>>(tA, tmp, 32 * D_);
    }
    ln_kernel<<<32, 256, 0, stream>>>(tA, tA, enc_ng, enc_nb, D_);
    add_bcast_kernel<<<blocks(32 * D_), TPB, 0, stream>>>(tA, dec_pos, 32 * D_, N_ * D_);

    ln_kernel<<<32, 256, 0, stream>>>(tA, tB, dec_aln_g, dec_aln_b, D_);
    gemm_wmma_kernel<<<gemm_grid(32, 3 * D_), 256, 0, stream>>>(tB, dec_wqkv, qkv, nullptr, 32, 3 * D_, D_, 0);
    attn_kernel<<<B_ * HEADS_, 256, 0, stream>>>(qkv, attnout);
    gemm_wmma_kernel<<<gemm_grid(32, D_), 256, 0, stream>>>(attnout, dec_wo, tmp, dec_bo, 32, D_, D_, 0);
    add_inplace_kernel<<<blocks(32 * D_), TPB, 0, stream>>>(tA, tmp, 32 * D_);
    ln_kernel<<<32, 256, 0, stream>>>(tA, tB, dec_fln_g, dec_fln_b, D_);
    gemm_wmma_kernel<<<gemm_grid(32, 3072), 256, 0, stream>>>(tB, dec_w1, ff1, dec_b1, 32, 3072, D_, 2);
    gemm_wmma_kernel<<<gemm_grid(32, D_), 256, 0, stream>>>(ff1, dec_w2, tmp, dec_b2, 32, D_, 3072, 0);
    add_inplace_kernel<<<blocks(32 * D_), TPB, 0, stream>>>(tA, tmp, 32 * D_);
    ln_kernel<<<32, 256, 0, stream>>>(tA, tA, dec_ng, dec_nb, D_);
    gemm_wmma_kernel<<<gemm_grid(32, 4096), 256, 0, stream>>>(tA, fin_w, lat, fin_b, 32, 4096, D_, 0);

    // ================= SS2D branch =================
    transpose_ksp_kernel<<<blocks((long long)B_ * L_ * C_), TPB, 0, stream>>>(in_ksp, ksp_t);
    gemm_wmma_kernel<<<gemm_grid(B_ * L_, 64), 256, 0, stream>>>(ksp_t, ss_in_w, xz, ss_in_b, B_ * L_, 64, C_, 0);
    dwconv_silu_kernel<<<blocks((long long)B_ * DI_ * L_), TPB, 0, stream>>>(xz, ss_conv_w, ss_conv_b, xs0);
    fill0_kernel<<<blocks((long long)B_ * L_ * DI_), TPB, 0, stream>>>(ybuf, B_ * L_ * DI_);
    ss_proj_kernel<<<blocks((long long)B_ * KDIR_ * L_), TPB, 0, stream>>>(
        xs0, ss_xproj, ss_dt_w, ss_dt_b, dts, Bmb, Cmb);
    ss_scan_kernel<<<B_ * KDIR_ * DI_, 256, 0, stream>>>(dts, xs0, Bmb, Cmb, ss_Alog, ss_D, ybuf);
    ss_lngate_kernel<<<blocks((long long)B_ * L_), TPB, 0, stream>>>(ybuf, xz, ss_ln_g, ss_ln_b, ygate);
    gemm_wmma_kernel<<<gemm_grid(B_ * L_, OUTCH_), 256, 0, stream>>>(
        ygate, ss_out_w, out_ss, ss_out_b, B_ * L_, OUTCH_, DI_, 0);

    // ================= conv head (im2col + WMMA) =================
    concat_kernel<<<blocks((long long)B_ * L_ * 56), TPB, 0, stream>>>(lat, in_imgs, out_ss, xc);
    repack_w_kernel<<<blocks(9 * 56 * 64), TPB, 0, stream>>>(head_w, wbuf, 64, 56);
    im2col_kernel<<<blocks((long long)B_ * L_ * 56 * 9), TPB, 0, stream>>>(xc, colbuf, 56);
    gemm_wmma_kernel<<<gemm_grid(B_ * L_, 64), 256, 0, stream>>>(colbuf, wbuf, h0, head_b, B_ * L_, 64, 504, 1);
    for (int i = 0; i < 3; ++i) {
        im2col_kernel<<<blocks((long long)B_ * L_ * 576), TPB, 0, stream>>>(h0, colbuf, 64);
        repack_w_kernel<<<blocks(9 * 64 * 64), TPB, 0, stream>>>(res_w1 + (size_t)i * 64 * 64 * 9, wbuf, 64, 64);
        gemm_wmma_kernel<<<gemm_grid(B_ * L_, 64), 256, 0, stream>>>(colbuf, wbuf, h1, res_b1 + i * 64, B_ * L_, 64, 576, 1);
        im2col_kernel<<<blocks((long long)B_ * L_ * 576), TPB, 0, stream>>>(h1, colbuf, 64);
        repack_w_kernel<<<blocks(9 * 64 * 64), TPB, 0, stream>>>(res_w2 + (size_t)i * 64 * 64 * 9, wbuf, 64, 64);
        gemm_wmma_kernel<<<gemm_grid(B_ * L_, 64), 256, 0, stream>>>(colbuf, wbuf, h2, res_b2 + i * 64, B_ * L_, 64, 576, 0);
        resadd_lrelu_kernel<<<blocks((long long)B_ * L_ * 64), TPB, 0, stream>>>(h0, h2, B_ * L_ * 64);
    }
    im2col_kernel<<<blocks((long long)B_ * L_ * 576), TPB, 0, stream>>>(h0, colbuf, 64);
    repack_w_kernel<<<blocks(9 * 64 * 2), TPB, 0, stream>>>(tail_w, wbuf, 2, 64);
    gemm_wmma_kernel<<<gemm_grid(B_ * L_, 2), 256, 0, stream>>>(colbuf, wbuf, xri, tail_b, B_ * L_, 2, 576, 0);

    // ================= FFT data consistency =================
    kprep_kernel<<<blocks((long long)B_ * 16 * L_), TPB, 0, stream>>>(kbuf, sens, xri);
    fft128_kernel<<<B_ * 16 * H_, 128, 0, stream>>>(kbuf, 0, -1.f);
    fft128_kernel<<<B_ * 16 * W_, 128, 0, stream>>>(kbuf, 1, -1.f);
    dc_kernel<<<blocks((long long)B_ * 16 * L_), TPB, 0, stream>>>(kbuf, in_ksp, maskp, dc_alpha);
    fft128_kernel<<<B_ * 16 * H_, 128, 0, stream>>>(kbuf, 0, +1.f);
    fft128_kernel<<<B_ * 16 * W_, 128, 0, stream>>>(kbuf, 1, +1.f);
    combine_kernel<<<blocks((long long)B_ * L_), TPB, 0, stream>>>(kbuf, sens, out);
}
